// RQVAE_38431367364810
// MI455X (gfx1250) — compile-verified, matrix operand order checked by reading, not run on hardware
//
#include <hip/hip_runtime.h>

// ---------------------------------------------------------------------------
// RQ-VAE forward loss for MI455X (gfx1250, wave32, WMMA + double-buffered
// async global->LDS staging). bf16 WMMA GEMMs (f32 accumulate); f32 VQ argmin.
// ---------------------------------------------------------------------------

typedef __bf16        v16bf __attribute__((ext_vector_type(16)));
typedef float         v8f   __attribute__((ext_vector_type(8)));
typedef unsigned int  v4u   __attribute__((ext_vector_type(4)));
typedef float         v4f   __attribute__((ext_vector_type(4)));

#if defined(__has_builtin)
#  if __has_builtin(__builtin_amdgcn_cvt_pk_bf16_f32)
#    define HAVE_CVT_PK_BF16 1
#  endif
#endif

static __device__ __forceinline__ unsigned short f2bf(float f) {
    unsigned u = __float_as_uint(f);
    unsigned r = (u + 0x7FFFu + ((u >> 16) & 1u)) >> 16;  // round-to-nearest-even
    return (unsigned short)r;
}
static __device__ __forceinline__ unsigned pack2(float a, float b) {
#ifdef HAVE_CVT_PK_BF16
    typedef __bf16 v2bf __attribute__((ext_vector_type(2)));
    v2bf r = __builtin_amdgcn_cvt_pk_bf16_f32(a, b);
    return __builtin_bit_cast(unsigned, r);
#else
    return (unsigned)f2bf(a) | ((unsigned)f2bf(b) << 16);
#endif
}

// Async global->LDS tile staging (ASYNCcnt-tracked; overlaps with WMMA).
// LDS operand: low 32 bits of generic shared pointer == DS byte address.
#define ASYNC_B128(ldsoff_u32, gaddr_u64)                                  \
    asm volatile("global_load_async_to_lds_b128 %0, %1, off"               \
                 :: "v"(ldsoff_u32), "v"(gaddr_u64) : "memory")

static __device__ __forceinline__ void wait_async0() {
    asm volatile("s_wait_asynccnt 0" ::: "memory");
}

struct Frag2 { v4u lo, hi; };

// ---------------------------------------------------------------------------
// Weight f32 -> bf16 conversion
// ---------------------------------------------------------------------------
__global__ __launch_bounds__(256) void cvt_kernel(const float* __restrict__ s,
                                                  unsigned short* __restrict__ d,
                                                  int n) {
    int i = blockIdx.x * 256 + threadIdx.x;
    if (i < n) d[i] = f2bf(s[i]);
}

// ---------------------------------------------------------------------------
// Tiled WMMA GEMM:  C[M x DOUT] = act(A[M x K] * W^T + bias)
//   W is torch-layout [DOUT, K]; B fragment loads are contiguous 16B chunks
//   per lane, matching the ISA B layout (lane=col, K split
//   {h*8..h*8+7, 16+h*8..16+h*8+7} for h = lane>>4).
// Block tile 128x64, BK=32, 8 waves (4 in M, 2 in N), 2x2 16x16 frags/wave.
// Double-buffered LDS: tile k+32 is staged (async for bf16 A / B, VGPR
// convert for f32 A) while WMMA consumes tile k; one wait+barrier per step.
// MODE 0: ReLU -> bf16 store.  MODE 1: linear -> f32 store.
// MODE 2: fused recon loss: sum_n (val - xref)^2 accumulated per row.
// ---------------------------------------------------------------------------
template <int MODE, bool AF32>
__global__ __launch_bounds__(256) void gemm_kernel(
    const void* __restrict__ Aptr, const unsigned short* __restrict__ Wb,
    const float* __restrict__ bias, int K, int DOUT,
    unsigned short* __restrict__ outb, float* __restrict__ outf,
    const float* __restrict__ xref, float* __restrict__ rowloss) {

    __shared__ __align__(16) unsigned short lds_a[2][128 * 48];
    __shared__ __align__(16) unsigned short lds_b[2][64 * 48];

    const int tid  = threadIdx.x;
    const int lane = tid & 31;
    const int wave = tid >> 5;
    const int wm   = wave & 3;   // wave position in M (0..3)
    const int wn   = wave >> 2;  // wave position in N (0..1)
    const int row0 = blockIdx.y * 128;
    const int col0 = blockIdx.x * 64;
    const int m    = lane & 15;
    const int h    = lane >> 4;

    v8f acc[2][2] = {};

    const int ar = tid >> 1, ah = tid & 1;  // A staging: row, 16-elem half
    const int bn = tid >> 2, bq = tid & 3;  // B staging: row n, 8-elem quarter
    const bool b_ok = (col0 + bn) < DOUT;

    unsigned a_lds0[2], a_lds1[2], b_lds[2];
    #pragma unroll
    for (int b = 0; b < 2; ++b) {
        a_lds0[b] = (unsigned)(uintptr_t)&lds_a[b][ar * 48 + ah * 16];
        a_lds1[b] = (unsigned)(uintptr_t)&lds_a[b][ar * 48 + ah * 16 + 8];
        b_lds[b]  = (unsigned)(uintptr_t)&lds_b[b][bn * 48 + bq * 8];
    }

    // Stage one 128x32 A tile + 64x32 B tile for K offset k0 into buffer buf.
    auto stage = [&](int k0, int buf) {
        if (AF32) {
            const float* A = (const float*)Aptr;
            const v4f* s = (const v4f*)(A + (size_t)(row0 + ar) * K + k0 + ah * 16);
            v4f f0 = s[0], f1 = s[1], f2 = s[2], f3 = s[3];
            v4u q0, q1;
            q0.x = pack2(f0.x, f0.y); q0.y = pack2(f0.z, f0.w);
            q0.z = pack2(f1.x, f1.y); q0.w = pack2(f1.z, f1.w);
            q1.x = pack2(f2.x, f2.y); q1.y = pack2(f2.z, f2.w);
            q1.z = pack2(f3.x, f3.y); q1.w = pack2(f3.z, f3.w);
            *(v4u*)&lds_a[buf][ar * 48 + ah * 16]     = q0;
            *(v4u*)&lds_a[buf][ar * 48 + ah * 16 + 8] = q1;
        } else {
            const unsigned short* A = (const unsigned short*)Aptr;
            unsigned long long g =
                (unsigned long long)(uintptr_t)(A + (size_t)(row0 + ar) * K + k0 + ah * 16);
            ASYNC_B128(a_lds0[buf], g);
            ASYNC_B128(a_lds1[buf], g + 16ull);
        }
        if (b_ok) {
            unsigned long long g =
                (unsigned long long)(uintptr_t)(Wb + (size_t)(col0 + bn) * K + k0 + bq * 8);
            ASYNC_B128(b_lds[buf], g);
        }
    };

    // Prologue: fill buffer 0 with the first tile.
    stage(0, 0);
    wait_async0();
    __syncthreads();

    int cur = 0;
    for (int k0 = 0; k0 < K; k0 += 32) {
        // Prefetch next tile into the alternate buffer while computing.
        if (k0 + 32 < K) stage(k0 + 32, cur ^ 1);

        // ---- fragment loads per ISA wave32 layout ----
        v16bf afrag[2], bfrag[2];
        #pragma unroll
        for (int rb = 0; rb < 2; ++rb) {
            int r = wm * 32 + rb * 16 + m;
            Frag2 f;
            f.lo = *(const v4u*)&lds_a[cur][r * 48 + h * 8];       // K=h*8..
            f.hi = *(const v4u*)&lds_a[cur][r * 48 + 16 + h * 8];  // K=16+h*8..
            afrag[rb] = __builtin_bit_cast(v16bf, f);
        }
        #pragma unroll
        for (int cbi = 0; cbi < 2; ++cbi) {
            int n = wn * 32 + cbi * 16 + m;
            Frag2 f;
            f.lo = *(const v4u*)&lds_b[cur][n * 48 + h * 8];
            f.hi = *(const v4u*)&lds_b[cur][n * 48 + 16 + h * 8];
            bfrag[cbi] = __builtin_bit_cast(v16bf, f);
        }
        #pragma unroll
        for (int rb = 0; rb < 2; ++rb)
            #pragma unroll
            for (int cbi = 0; cbi < 2; ++cbi)
                acc[rb][cbi] = __builtin_amdgcn_wmma_f32_16x16x32_bf16(
                    false, afrag[rb], false, bfrag[cbi],
                    (short)0, acc[rb][cbi], false, false);

        // Join: own async copies done, then workgroup-wide visibility.
        wait_async0();
        __syncthreads();
        cur ^= 1;
    }

    // ---- epilogue ----
    #pragma unroll
    for (int rb = 0; rb < 2; ++rb) {
        #pragma unroll
        for (int cbi = 0; cbi < 2; ++cbi) {
            int rbase = row0 + wm * 32 + rb * 16 + h * 8;  // rows rbase..rbase+7
            int gcol  = col0 + wn * 32 + cbi * 16 + m;
            float bz = (gcol < DOUT) ? bias[gcol] : 0.0f;
            if (MODE == 2) {
                // fused recon loss: per-row sum of squared error across 16 cols
                #pragma unroll
                for (int v = 0; v < 8; ++v) {
                    size_t idx = (size_t)(rbase + v) * DOUT + gcol;
                    float d = acc[rb][cbi][v] + bz - xref[idx];
                    float s = d * d;
                    #pragma unroll
                    for (int off = 1; off < 16; off <<= 1)
                        s += __shfl_xor(s, off, 32);
                    if (m == 0) atomicAdd(&rowloss[rbase + v], s);
                }
            } else if (gcol < DOUT) {
                #pragma unroll
                for (int v = 0; v < 8; ++v) {
                    float val = acc[rb][cbi][v] + bz;
                    size_t idx = (size_t)(rbase + v) * DOUT + gcol;
                    if (MODE == 0) {
                        val = val > 0.0f ? val : 0.0f;  // ReLU
                        outb[idx] = f2bf(val);
                    } else {
                        outf[idx] = val;
                    }
                }
            }
        }
    }
}

// ---------------------------------------------------------------------------
// Residual VQ: 3 levels, 256x32 f32 codebook per level staged in LDS.
// One thread per row; all lanes scan the same codeword -> LDS broadcast.
// Writes rowloss = 1.25 * sum_l ||residual_l||^2 and bf16 z_hat.
// ---------------------------------------------------------------------------
__global__ __launch_bounds__(256) void rq_kernel(
    const float* __restrict__ z, const float* __restrict__ cb0,
    const float* __restrict__ cb1, const float* __restrict__ cb2,
    unsigned int* __restrict__ zhat, float* __restrict__ rowloss) {

    __shared__ __align__(16) float cbs[256 * 32];
    const int row = blockIdx.x * 256 + threadIdx.x;
    const float* zr = z + (size_t)row * 32;
    float r[32];
    #pragma unroll
    for (int k = 0; k < 32; ++k) r[k] = zr[k];

    float loss = 0.0f;
    const float* tabs[3] = {cb0, cb1, cb2};
    for (int lvl = 0; lvl < 3; ++lvl) {
        __syncthreads();
        for (int i = threadIdx.x; i < 256 * 32; i += 256) cbs[i] = tabs[lvl][i];
        __syncthreads();
        float best = 3.4e38f;
        int   bj   = 0;
        for (int j = 0; j < 256; ++j) {
            float d = 0.0f;
            #pragma unroll
            for (int k = 0; k < 32; ++k) {
                float t = r[k] - cbs[j * 32 + k];
                d += t * t;
            }
            if (d < best) { best = d; bj = j; }  // first-min, matches argmin
        }
        float l2 = 0.0f;
        #pragma unroll
        for (int k = 0; k < 32; ++k) {
            r[k] -= cbs[bj * 32 + k];
            l2 += r[k] * r[k];
        }
        loss += l2;  // ||res_l - cw_l||^2 == ||residual_{l+1}||^2
    }
    rowloss[row] = 1.25f * loss;  // (1 + BETA) * commitment terms
    #pragma unroll
    for (int k = 0; k < 16; ++k)  // z_hat = sum of codewords, packed bf16x2
        zhat[(size_t)row * 16 + k] = pack2(zr[2 * k] - r[2 * k],
                                           zr[2 * k + 1] - r[2 * k + 1]);
}

// ---------------------------------------------------------------------------
// Final mean reduction (single block, deterministic)
// ---------------------------------------------------------------------------
__global__ __launch_bounds__(256) void reduce_kernel(const float* __restrict__ rl,
                                                     float* __restrict__ out, int n) {
    __shared__ float sm[256];
    float a = 0.0f;
    for (int i = threadIdx.x; i < n; i += 256) a += rl[i];
    sm[threadIdx.x] = a;
    __syncthreads();
    for (int s = 128; s > 0; s >>= 1) {
        if (threadIdx.x < s) sm[threadIdx.x] += sm[threadIdx.x + s];
        __syncthreads();
    }
    if (threadIdx.x == 0) out[0] = sm[0] / (float)n;
}

// ---------------------------------------------------------------------------
// Host side
// Input order (setup_inputs dict, tuples flattened):
//   0: x [N*768]      1: temperature (unused)
//   2-5:  enc_ws      6-9:  enc_bs
//   10-13: dec_ws     14-17: dec_bs
//   18-20: codebooks [256*32] each
// ---------------------------------------------------------------------------
extern "C" void kernel_launch(void* const* d_in, const int* in_sizes, int n_in,
                              void* d_out, int out_size, void* d_ws, size_t ws_size,
                              hipStream_t stream) {
    (void)in_sizes; (void)n_in; (void)out_size; (void)ws_size;
    const int N = 131072;

    const float* x = (const float*)d_in[0];
    const float* enc_w[4] = {(const float*)d_in[2],  (const float*)d_in[3],
                             (const float*)d_in[4],  (const float*)d_in[5]};
    const float* enc_b[4] = {(const float*)d_in[6],  (const float*)d_in[7],
                             (const float*)d_in[8],  (const float*)d_in[9]};
    const float* dec_w[4] = {(const float*)d_in[10], (const float*)d_in[11],
                             (const float*)d_in[12], (const float*)d_in[13]};
    const float* dec_b[4] = {(const float*)d_in[14], (const float*)d_in[15],
                             (const float*)d_in[16], (const float*)d_in[17]};
    const float* cb0 = (const float*)d_in[18];
    const float* cb1 = (const float*)d_in[19];
    const float* cb2 = (const float*)d_in[20];

    const int encK[4] = {768, 512, 256, 128};
    const int encD[4] = {512, 256, 128, 32};
    const int decK[4] = {32, 128, 256, 512};
    const int decD[4] = {128, 256, 512, 768};

    char* ws = (char*)d_ws;
    size_t off = 0;
    auto take = [&](size_t bytes) -> void* {
        void* p = ws + off;
        off = (off + bytes + 255) & ~(size_t)255;
        return p;
    };

    unsigned short* wencb[4];
    unsigned short* wdecb[4];
    for (int i = 0; i < 4; ++i)
        wencb[i] = (unsigned short*)take((size_t)encD[i] * encK[i] * 2);
    for (int i = 0; i < 4; ++i)
        wdecb[i] = (unsigned short*)take((size_t)decD[i] * decK[i] * 2);
    unsigned short* actA    = (unsigned short*)take((size_t)N * 512 * 2);
    unsigned short* actB    = (unsigned short*)take((size_t)N * 512 * 2);
    float*          zbuf    = (float*)take((size_t)N * 32 * 4);
    unsigned int*   zhat    = (unsigned int*)take((size_t)N * 32 * 2);
    float*          rowloss = (float*)take((size_t)N * 4);

    // Weights -> bf16 (deterministic, every call)
    for (int i = 0; i < 4; ++i) {
        int n = encD[i] * encK[i];
        cvt_kernel<<<(n + 255) / 256, 256, 0, stream>>>(enc_w[i], wencb[i], n);
    }
    for (int i = 0; i < 4; ++i) {
        int n = decD[i] * decK[i];
        cvt_kernel<<<(n + 255) / 256, 256, 0, stream>>>(dec_w[i], wdecb[i], n);
    }

    dim3 blk(256);
    const int GY = N / 128;

    // Encoder MLP
    gemm_kernel<0, true ><<<dim3(512 / 64, GY), blk, 0, stream>>>(
        x,    wencb[0], enc_b[0], 768, 512, actA, nullptr, nullptr, nullptr);
    gemm_kernel<0, false><<<dim3(256 / 64, GY), blk, 0, stream>>>(
        actA, wencb[1], enc_b[1], 512, 256, actB, nullptr, nullptr, nullptr);
    gemm_kernel<0, false><<<dim3(128 / 64, GY), blk, 0, stream>>>(
        actB, wencb[2], enc_b[2], 256, 128, actA, nullptr, nullptr, nullptr);
    gemm_kernel<1, false><<<dim3(1, GY), blk, 0, stream>>>(
        actA, wencb[3], enc_b[3], 128, 32, nullptr, zbuf, nullptr, nullptr);

    // Residual vector quantization (writes rowloss + bf16 z_hat)
    rq_kernel<<<N / 256, blk, 0, stream>>>(zbuf, cb0, cb1, cb2, zhat, rowloss);

    // Decoder MLP; last layer fuses recon loss against x into rowloss
    gemm_kernel<0, false><<<dim3(128 / 64, GY), blk, 0, stream>>>(
        zhat, wdecb[0], dec_b[0], 32, 128, actA, nullptr, nullptr, nullptr);
    gemm_kernel<0, false><<<dim3(256 / 64, GY), blk, 0, stream>>>(
        actA, wdecb[1], dec_b[1], 128, 256, actB, nullptr, nullptr, nullptr);
    gemm_kernel<0, false><<<dim3(512 / 64, GY), blk, 0, stream>>>(
        actB, wdecb[2], dec_b[2], 256, 512, actA, nullptr, nullptr, nullptr);
    gemm_kernel<2, false><<<dim3(768 / 64, GY), blk, 0, stream>>>(
        actA, wdecb[3], dec_b[3], 512, 768, nullptr, nullptr, x, rowloss);

    // mean over rows -> scalar
    reduce_kernel<<<1, blk, 0, stream>>>(rowloss, (float*)d_out, N);
}
